// InstantNGPModel_11587821765209
// MI455X (gfx1250) — compile-verified, matrix operand order checked by reading, not run on hardware
//
#include <hip/hip_runtime.h>
#include <hip/hip_bf16.h>
#include <stdint.h>

typedef _Float16 half_t;
typedef __attribute__((ext_vector_type(16))) _Float16 v16h;
typedef __attribute__((ext_vector_type(8)))  float    v8f;

#define TPB 128   // 4 wave32 per block, 128 points per block

// Weight segment offsets inside d_ws (in halves), transposed [n][Kpad] layout.
#define OFF_W0 0        // 64 x 32
#define OFF_W1 2048     // 64 x 64
#define OFF_W2 6144     // 16 x 64
#define OFF_C0 7168     // 64 x 32
#define OFF_C1 9216     // 64 x 64
#define OFF_C2 13312    // 64 x 64
#define OFF_C3 17408    // 16 x 64
#define WS_HALVES 18432 // total = 36864 bytes, L2-resident

// ---------------------------------------------------------------------------
// Fragment loader: 16-bit A-layout (16x32 tile) from a row-major f16 matrix
// (LDS activations or global weights) with `stride` halves per row.
// Per ISA 7.12.2:
//   lanes 0-15 : row = row0+lane,    K-octet base 0 (v0..3) / 16 (v4..7)
//   lanes 16-31: row = row0+lane-16, K-octet base 8 (v0..3) / 24 (v4..7)
// ---------------------------------------------------------------------------
__device__ __forceinline__ v16h load_frag(const half_t* p, int row0, int k0,
                                          int stride, int lane) {
  const half_t* base = p + (row0 + (lane & 15)) * stride + k0 + ((lane >> 4) << 3);
  union { v16h v; uint32_t u[8]; } f;
#pragma unroll
  for (int i = 0; i < 4; ++i) f.u[i]     = *(const uint32_t*)(base + 2 * i);
#pragma unroll
  for (int i = 0; i < 4; ++i) f.u[4 + i] = *(const uint32_t*)(base + 16 + 2 * i);
  return f.v;
}

// Dense layer: out[Mx(NT*16)] = act(in[Mx(KT*32)] * W^T + bias).
// A fragments (LDS) hoisted into registers; B fragments stream from L2.
template <int NT, int KT, bool RELU>
__device__ __forceinline__ void mlp_layer(const half_t* in, int inS,
                                          const half_t* __restrict__ wT, // [N][KT*32] global
                                          const float* __restrict__ bias,
                                          half_t* out, int outS,
                                          int wv, int lane) {
  v16h a[2][KT];
#pragma unroll
  for (int mt = 0; mt < 2; ++mt)
#pragma unroll
    for (int kt = 0; kt < KT; ++kt)
      a[mt][kt] = load_frag(in, wv * 32 + mt * 16, kt * 32, inS, lane);

  const int col = lane & 15;
  const int rg  = (lane >> 4) << 3;
#pragma unroll
  for (int nt = 0; nt < NT; ++nt) {
    v16h b[KT];
#pragma unroll
    for (int kt = 0; kt < KT; ++kt)
      b[kt] = load_frag(wT, nt * 16, kt * 32, KT * 32, lane);
    const float bv = bias[nt * 16 + col];
#pragma unroll
    for (int mt = 0; mt < 2; ++mt) {
      v8f c = {};
#pragma unroll
      for (int kt = 0; kt < KT; ++kt)
        c = __builtin_amdgcn_wmma_f32_16x16x32_f16(false, a[mt][kt], false,
                                                   b[kt], (short)0, c,
                                                   false, false);
      const int m0 = wv * 32 + mt * 16;
#pragma unroll
      for (int i = 0; i < 8; ++i) {
        float v = c[i] + bv;
        if (RELU) v = fmaxf(v, 0.f);
        out[(m0 + rg + i) * outS + nt * 16 + col] = (half_t)v;
      }
    }
  }
}

// ---------------------------------------------------------------------------
// Prep kernel: convert all weights once to f16 transposed [n][Kpad] in d_ws.
// ---------------------------------------------------------------------------
__device__ __forceinline__ void cvt_kpad(half_t* dst, const float* src,
                                         int Kact, int Nn, int Kpad,
                                         int tid, int stride) {
  for (int i = tid; i < Nn * Kpad; i += stride) {
    const int n = i / Kpad, k = i - n * Kpad;
    dst[i] = (k < Kact) ? (half_t)src[k * Nn + n] : (half_t)0.f;
  }
}
__device__ __forceinline__ void cvt_npad(half_t* dst, const float* src,
                                         int K, int Nact, int Npad,
                                         int tid, int stride) {
  for (int i = tid; i < Npad * K; i += stride) {
    const int n = i / K, k = i - n * K;
    dst[i] = (n < Nact) ? (half_t)src[k * Nact + n] : (half_t)0.f;
  }
}

__global__ void prep_weights(const float* __restrict__ dw0,
                             const float* __restrict__ dw1,
                             const float* __restrict__ dw2,
                             const float* __restrict__ cw0,
                             const float* __restrict__ cw1,
                             const float* __restrict__ cw2,
                             const float* __restrict__ cw3,
                             half_t* __restrict__ ws) {
  const int tid    = blockIdx.x * blockDim.x + threadIdx.x;
  const int stride = gridDim.x * blockDim.x;
  cvt_kpad(ws + OFF_W0, dw0, 32, 64, 32, tid, stride);
  cvt_kpad(ws + OFF_W1, dw1, 64, 64, 64, tid, stride);
  cvt_npad(ws + OFF_W2, dw2, 64, 16, 16, tid, stride);
  cvt_kpad(ws + OFF_C0, cw0, 31, 64, 32, tid, stride);
  cvt_kpad(ws + OFF_C1, cw1, 64, 64, 64, tid, stride);
  cvt_kpad(ws + OFF_C2, cw2, 64, 64, 64, tid, stride);
  cvt_npad(ws + OFF_C3, cw3, 64, 3, 16, tid, stride);
}

// ---------------------------------------------------------------------------
// Fused NGP forward.
// ---------------------------------------------------------------------------
__global__ __launch_bounds__(TPB) void ngp_fused(
    const float* __restrict__ pos, const float* __restrict__ dir,
    const float* __restrict__ emb,
    const half_t* __restrict__ ws,                     // transposed f16 weights
    const float* __restrict__ db0, const float* __restrict__ db1,
    const float* __restrict__ db2,
    const float* __restrict__ cb0, const float* __restrict__ cb1,
    const float* __restrict__ cb2, const float* __restrict__ cb3,
    float* __restrict__ out, int N) {
  __shared__ half_t bufA[TPB * 64];
  __shared__ half_t bufB[TPB * 64];
  __shared__ half_t cbuf[TPB * 32];   // geo(15) | SH(16) | pad(1)
  __shared__ float  sigs[TPB];

  const int tid   = threadIdx.x;
  const int lane  = tid & 31;
  const int wv    = tid >> 5;
  const int gbase = blockIdx.x * TPB;

  // Warm L2/WGP$ with the weight region (global_prefetch_b8, gfx1250 path).
  __builtin_prefetch(ws + tid * (WS_HALVES / TPB), 0, 1);

  // ---- hash-grid encode (per point -> bufA row, f16) --------------------
  int pidx = gbase + tid;
  if (pidx >= N) pidx = N - 1;
  float pn[3];
#pragma unroll
  for (int c = 0; c < 3; ++c) {
    const float v = (pos[pidx * 3 + c] + 1.f) * 0.5f;   // AABB [-1,1] -> [0,1]
    pn[c] = fminf(fmaxf(v, 0.f), 1.f);
  }
  const float2* e2 = (const float2*)emb;
  for (int lvl = 0; lvl < 16; ++lvl) {                   // uniform loop (SALU)
    int res = 16 << lvl; if (res > 512) res = 512;
    int lg = 12 + 3 * lvl; if (lg > 19) lg = 19;         // sizes are powers of 2
    const uint32_t mask = (1u << lg) - 1u;
    const uint32_t off  = (lvl == 0) ? 0u
                        : (lvl == 1) ? 4096u
                        : (lvl == 2) ? 36864u
                        : 299008u + (uint32_t)(lvl - 3) * 524288u;
    const float s = (float)(res - 1);
    const uint32_t rm1 = (uint32_t)(res - 1);
    float w[3]; uint32_t pg[3];
#pragma unroll
    for (int c = 0; c < 3; ++c) {
      const float psc = pn[c] * s;
      float fl = floorf(psc);
      fl = fminf(fmaxf(fl, 0.f), s);
      w[c]  = psc - fl;
      pg[c] = (uint32_t)fl;
    }
    float f0 = 0.f, f1 = 0.f;
#pragma unroll
    for (int cn = 0; cn < 8; ++cn) {                     // 8 corner gathers (L2)
      const uint32_t dx = (cn >> 2) & 1u, dy = (cn >> 1) & 1u, dz = cn & 1u;
      uint32_t cx = pg[0] + dx; cx = cx > rm1 ? rm1 : cx;
      uint32_t cy = pg[1] + dy; cy = cy > rm1 ? rm1 : cy;
      uint32_t cz = pg[2] + dz; cz = cz > rm1 ? rm1 : cz;
      const uint32_t h = (cx * 73856093u + cy * 19349663u + cz * 83492791u) & mask;
      const float2 e = e2[h + off];
      const float wt = (dx ? w[0] : 1.f - w[0]) *
                       (dy ? w[1] : 1.f - w[1]) *
                       (dz ? w[2] : 1.f - w[2]);
      f0 += wt * e.x; f1 += wt * e.y;
    }
    bufA[tid * 64 + 2 * lvl]     = (half_t)f0;
    bufA[tid * 64 + 2 * lvl + 1] = (half_t)f1;
  }

  // ---- SH degree-3 encode -> cbuf cols 15..30, col 31 = 0 ---------------
  {
    const float vx = dir[pidx * 3], vy = dir[pidx * 3 + 1], vz = dir[pidx * 3 + 2];
    const float inv = 1.f / sqrtf(vx * vx + vy * vy + vz * vz);
    const float x = vx * inv, y = vy * inv, z = vz * inv;
    const float x2 = x * x, y2 = y * y, z2 = z * z;
    const float sh[16] = {
      0.28209479177387814f,
      -0.48860251190291987f * y,
      0.48860251190291987f * z,
      -0.48860251190291987f * x,
      1.0925484305920792f * x * y,
      -1.0925484305920792f * y * z,
      0.31539156525252005f * (2.f * z2 - x2 - y2),
      -1.0925484305920792f * x * z,
      0.5462742152960396f * (x2 - y2),
      -0.5900435899266435f * y * (3.f * x2 - y2),
      2.890611442640554f * x * y * z,
      -0.4570457994644658f * y * (4.f * z2 - x2 - y2),
      0.3731763325901154f * z * (2.f * z2 - 3.f * x2 - 3.f * y2),
      -0.4570457994644658f * x * (4.f * z2 - x2 - y2),
      1.445305721320277f * z * (x2 - y2),
      -0.5900435899266435f * x * (x2 - 3.f * y2)};
#pragma unroll
    for (int j = 0; j < 16; ++j) cbuf[tid * 32 + 15 + j] = (half_t)sh[j];
    cbuf[tid * 32 + 31] = (half_t)0.f;
  }
  __syncthreads();

  // ---- density MLP: 32 -> 64 -> 64 -> 16 (WMMA f16, f32 accum) ----------
  mlp_layer<4, 1, true>(bufA, 64, ws + OFF_W0, db0, bufB, 64, wv, lane);
  __syncthreads();
  mlp_layer<4, 2, true>(bufB, 64, ws + OFF_W1, db1, bufA, 64, wv, lane);
  __syncthreads();
  {  // L2: no activation; col0 -> sigma (f32), cols 1..15 -> geo (cbuf 0..14)
    v16h b0 = load_frag(ws + OFF_W2, 0, 0, 64, lane);
    v16h b1 = load_frag(ws + OFF_W2, 0, 32, 64, lane);
    const int col = lane & 15;
    const int rg  = (lane >> 4) << 3;
    const float bv = db2[col];
#pragma unroll
    for (int mt = 0; mt < 2; ++mt) {
      const int m0 = wv * 32 + mt * 16;
      v16h a0 = load_frag(bufA, m0, 0, 64, lane);
      v16h a1 = load_frag(bufA, m0, 32, 64, lane);
      v8f c = {};
      c = __builtin_amdgcn_wmma_f32_16x16x32_f16(false, a0, false, b0,
                                                 (short)0, c, false, false);
      c = __builtin_amdgcn_wmma_f32_16x16x32_f16(false, a1, false, b1,
                                                 (short)0, c, false, false);
#pragma unroll
      for (int i = 0; i < 8; ++i) {
        const int row = m0 + rg + i;
        const float v = c[i] + bv;
        if (col == 0) sigs[row] = v;
        else          cbuf[row * 32 + (col - 1)] = (half_t)v;
      }
    }
  }
  __syncthreads();

  // ---- color MLP: 31(+pad) -> 64 -> 64 -> 64 -> 3 -----------------------
  mlp_layer<4, 1, true>(cbuf, 32, ws + OFF_C0, cb0, bufB, 64, wv, lane);
  __syncthreads();
  mlp_layer<4, 2, true>(bufB, 64, ws + OFF_C1, cb1, bufA, 64, wv, lane);
  __syncthreads();
  mlp_layer<4, 2, true>(bufA, 64, ws + OFF_C2, cb2, bufB, 64, wv, lane);
  __syncthreads();
  {  // L3: sigmoid, cols 0..2 -> rgb
    v16h b0 = load_frag(ws + OFF_C3, 0, 0, 64, lane);
    v16h b1 = load_frag(ws + OFF_C3, 0, 32, 64, lane);
    const int col = lane & 15;
    const int rg  = (lane >> 4) << 3;
#pragma unroll
    for (int mt = 0; mt < 2; ++mt) {
      const int m0 = wv * 32 + mt * 16;
      v16h a0 = load_frag(bufB, m0, 0, 64, lane);
      v16h a1 = load_frag(bufB, m0, 32, 64, lane);
      v8f c = {};
      c = __builtin_amdgcn_wmma_f32_16x16x32_f16(false, a0, false, b0,
                                                 (short)0, c, false, false);
      c = __builtin_amdgcn_wmma_f32_16x16x32_f16(false, a1, false, b1,
                                                 (short)0, c, false, false);
      if (col < 3) {
        const float bv = cb3[col];
#pragma unroll
        for (int i = 0; i < 8; ++i) {
          const int row = m0 + rg + i;
          const int gp  = gbase + row;
          if (gp < N) {
            const float v = c[i] + bv;
            out[gp * 4 + 1 + col] = 1.f / (1.f + __expf(-v));
          }
        }
      }
    }
  }
  // sigma
  {
    const int gp = gbase + tid;
    if (gp < N) out[gp * 4] = sigs[tid];
  }
}

extern "C" void kernel_launch(void* const* d_in, const int* in_sizes, int n_in,
                              void* d_out, int out_size, void* d_ws, size_t ws_size,
                              hipStream_t stream) {
  (void)n_in; (void)ws_size; (void)out_size;
  const float* pos = (const float*)d_in[0];
  const float* dir = (const float*)d_in[1];
  const float* emb = (const float*)d_in[2];
  const float* dw0 = (const float*)d_in[3];
  const float* db0 = (const float*)d_in[4];
  const float* dw1 = (const float*)d_in[5];
  const float* db1 = (const float*)d_in[6];
  const float* dw2 = (const float*)d_in[7];
  const float* db2 = (const float*)d_in[8];
  const float* cw0 = (const float*)d_in[9];
  const float* cb0 = (const float*)d_in[10];
  const float* cw1 = (const float*)d_in[11];
  const float* cb1 = (const float*)d_in[12];
  const float* cw2 = (const float*)d_in[13];
  const float* cb2 = (const float*)d_in[14];
  const float* cw3 = (const float*)d_in[15];
  const float* cb3 = (const float*)d_in[16];
  half_t* ws = (half_t*)d_ws;
  float* out = (float*)d_out;

  const int N = in_sizes[0] / 3;

  hipLaunchKernelGGL(prep_weights, dim3(32), dim3(128), 0, stream,
                     dw0, dw1, dw2, cw0, cw1, cw2, cw3, ws);

  const int grid = (N + TPB - 1) / TPB;
  hipLaunchKernelGGL(ngp_fused, dim3(grid), dim3(TPB), 0, stream,
                     pos, dir, emb, ws, db0, db1, db2,
                     cb0, cb1, cb2, cb3, out, N);
}
